// Pure_Coding_54546084659683
// MI455X (gfx1250) — matrix operand order repaired and checked by reading, not compiled, not run on hardware
//
#include <hip/hip_runtime.h>
#include <hip/hip_bf16.h>
#include <math.h>

// ---------------- problem constants ----------------
#define B_   768
#define L_   512
#define D_   15
#define F_   50
#define M_   32
#define KT_  5
#define CP   64               // padded channel count (multiple of 32 for WMMA K)
#define KFC  (L_ * CP)        // 32768 : FC reduction dim (padded)
#define NFC  512
#define KIN  25600            // original fc_w K = L_*F_

typedef __attribute__((ext_vector_type(16))) _Float16 v16h;
typedef __attribute__((ext_vector_type(8)))  _Float16 v8h;
typedef __attribute__((ext_vector_type(8)))  float    v8f;

// ---------------- weight prep: pack conv weights into B-fragment lane layout ----------------
// dst[((ct*nkc + kc)*32 + lane)*16 + e] = Wt[kc*32 + 16*(e>>3) + 8*(lane>>4) + (e&7)][ct*16 + (lane&15)]
// where Wt[tap*64 + ci][co] = w[co][ci][tap], zero-padded in ci/co.
__global__ __launch_bounds__(256)
void pack_conv_w_kernel(const float* __restrict__ w, _Float16* __restrict__ dst,
                        int cout, int cin, int ktaps)
{
    int nkc = ktaps * 2;
    int total = 4 * nkc * 32 * 16;
    int i = blockIdx.x * blockDim.x + threadIdx.x;
    if (i >= total) return;
    int e    = i & 15;
    int lane = (i >> 4) & 31;
    int rest = i >> 9;
    int kc   = rest % nkc;
    int Kg = kc * 32 + ((e >> 3) << 4) + ((lane >> 4) << 3) + (e & 7);      // row in (tap*64+ci) space
    int tap = Kg >> 6;
    int ci  = Kg & 63;
    int ct  = rest / nkc;
    int co  = (ct << 4) + (lane & 15);
    float v = 0.f;
    if (ci < cin && co < cout) v = w[((size_t)co * cin + ci) * ktaps + tap];
    dst[i] = (_Float16)v;
}

// ---------------- weight prep: pack fc weights (512, 25600) -> (nt, kc, lane, e) f16 ----------------
__global__ __launch_bounds__(256)
void pack_fc_w_kernel(const float* __restrict__ fcw, _Float16* __restrict__ dst)
{
    // total = 32 * 1024 * 32 * 16 = 16777216
    int i = blockIdx.x * blockDim.x + threadIdx.x;
    int e    = i & 15;
    int lane = (i >> 4) & 31;
    int rest = i >> 9;
    int kc   = rest & 1023;
    int nt   = rest >> 10;
    int Kg = kc * 32 + ((e >> 3) << 4) + ((lane >> 4) << 3) + (e & 7);      // K in (l*64 + f) padded space
    int l  = Kg >> 6;
    int f  = Kg & 63;
    int n  = (nt << 4) + (lane & 15);
    float v = 0.f;
    if (f < F_) v = fcw[(size_t)n * KIN + (size_t)l * F_ + f];
    dst[i] = (_Float16)v;
}

// ---------------- input pack: x (B,L) f32 -> act (B,L,64) f16 (ch0 = x, rest 0) ----------------
__global__ __launch_bounds__(256)
void pack_x_kernel(const float* __restrict__ x, _Float16* __restrict__ act)
{
    int i = blockIdx.x * blockDim.x + threadIdx.x;   // over B_*L_*CP
    int ci  = i & 63;
    int pos = i >> 6;
    act[i] = (ci == 0) ? (_Float16)x[pos] : (_Float16)0.f;
}

// ---------------- generic fused conv1d (+bias, +skip, +relu) via WMMA ----------------
// in/out/skip: (B, L, CP) f16 channels-last.  wpk: packed per-lane B fragments.
// grid.x = B_*(L_/16); block = 128 (4 waves, wave w owns one 16-channel cout tile)
template<int KTAPS, bool HAS_SKIP, bool RELU>
__global__ __launch_bounds__(128)
void conv_wmma_kernel(const _Float16* __restrict__ in, const _Float16* __restrict__ wpk,
                      const float* __restrict__ bias, const _Float16* __restrict__ skip,
                      _Float16* __restrict__ out, int coutUsed)
{
    __shared__ __align__(16) _Float16 win[20 * CP];   // positions ltile-2 .. ltile+17
    const int tile  = blockIdx.x;
    const int b     = tile >> 5;              // L_/16 == 32 tiles per batch row
    const int ltile = (tile & 31) << 4;
    const int tid   = threadIdx.x;

    // vectorized window fill: 20 rows x 64 ch = 160 x v8h, zero halo
    for (int ch = tid; ch < 160; ch += 128) {
        int p   = ch >> 3;
        int ci8 = (ch & 7) << 3;
        int gp  = ltile + p - 2;
        v8h v = {};
        if (gp >= 0 && gp < L_) v = *(const v8h*)(in + ((size_t)b * L_ + gp) * CP + ci8);
        *(v8h*)(win + p * CP + ci8) = v;
    }
    __syncthreads();

    const int lane    = tid & 31;
    const int ct      = tid >> 5;             // cout tile 0..3
    const int halfsel = lane >> 4;
    const int mrow    = lane & 15;            // A-matrix row for this lane
    constexpr int nkc  = KTAPS * 2;
    constexpr int roff = (5 - KTAPS) >> 1;    // re-center 1x1 conv inside the -2 halo

    v8f c = {};
    const _Float16* wp = wpk + (((size_t)ct * nkc) * 32 + lane) * 16;

    #pragma unroll
    for (int kc = 0; kc < nkc; ++kc) {
        const int tap = kc >> 1;
        const int cib = (kc & 1) << 5;
        const _Float16* al = win + (mrow + tap + roff) * CP + cib + 8 * halfsel;
        v16h a;
        *(v8h*)&a       = *(const v8h*)(al);        // K 0..7   of half-chunk
        *((v8h*)&a + 1) = *(const v8h*)(al + 16);   // K 16..23 of half-chunk
        v16h bm = *(const v16h*)(wp + (size_t)kc * 512);
        c = __builtin_amdgcn_wmma_f32_16x16x32_f16(false, a, false, bm,
                                                   (short)0, c, false, false);
    }

    // epilogue per C-fragment layout: col n = lane&15, rows r + 8*halfsel
    const int n  = lane & 15;
    const int co = (ct << 4) + n;
    const float bco = (co < coutUsed) ? bias[co] : 0.f;
    #pragma unroll
    for (int r = 0; r < 8; ++r) {
        const int m = r + (halfsel << 3);
        const size_t o = ((size_t)b * L_ + (ltile + m)) * CP + co;
        float v = c[r] + bco;
        if (HAS_SKIP) v += (float)skip[o];
        if (RELU) v = v > 0.f ? v : 0.f;
        out[o] = (_Float16)v;
    }
}

// ---------------- soft VQ: dist -> softmax prob -> xq ----------------
__global__ __launch_bounds__(256)
void code_assign_kernel(const _Float16* __restrict__ codes, const float* __restrict__ mean,
                        float* __restrict__ prob_out, _Float16* __restrict__ xq_out)
{
    __shared__ float smean[D_ * M_];
    __shared__ float sm2[M_];
    const int tid = threadIdx.x;
    for (int i = tid; i < D_ * M_; i += 256) smean[i] = mean[i];
    __syncthreads();
    if (tid < M_) {
        float s = 0.f;
        for (int d = 0; d < D_; ++d) { float v = smean[d * M_ + tid]; s += v * v; }
        sm2[tid] = s;
    }
    __syncthreads();

    const int idx  = blockIdx.x * 256 + tid;     // over B_*L_
    const size_t base = (size_t)idx * CP;

    float cd[D_];
    float c2 = 0.f;
    #pragma unroll
    for (int d = 0; d < D_; ++d) { cd[d] = (float)codes[base + d]; c2 += cd[d] * cd[d]; }

    float lg[M_];
    float mx = -3.4e38f;
    for (int m = 0; m < M_; ++m) {
        float cm = 0.f;
        #pragma unroll
        for (int d = 0; d < D_; ++d) cm += cd[d] * smean[d * M_ + m];
        float v = -(c2 + sm2[m] - 2.f * cm);     // SCALE = 1
        lg[m] = v;
        mx = fmaxf(mx, v);
    }
    float se = 0.f;
    for (int m = 0; m < M_; ++m) { lg[m] = __expf(lg[m] - mx); se += lg[m]; }
    const float inv = 1.f / se;

    float xq[D_];
    #pragma unroll
    for (int d = 0; d < D_; ++d) xq[d] = 0.f;
    for (int m = 0; m < M_; ++m) {
        float p = lg[m] * inv;
        prob_out[(size_t)idx * M_ + m] = p;
        #pragma unroll
        for (int d = 0; d < D_; ++d) xq[d] += p * smean[d * M_ + m];
    }
    #pragma unroll
    for (int d = 0; d < D_; ++d) xq_out[base + d] = (_Float16)xq[d];
    #pragma unroll
    for (int d = D_; d < CP; ++d) xq_out[base + d] = (_Float16)0.f;
}

// ---------------- FC GEMM 768x512x32768 + bias + tanh via WMMA ----------------
// grid = (48, 32), block = 32 (one wave per 16x16 output tile)
__global__ __launch_bounds__(32)
void fc_wmma_kernel(const _Float16* __restrict__ act, const _Float16* __restrict__ wfc,
                    const float* __restrict__ fcb, float* __restrict__ out)
{
    const int mt = blockIdx.x;
    const int nt = blockIdx.y;
    const int lane    = threadIdx.x;
    const int halfsel = lane >> 4;
    const int mrow    = lane & 15;

    const _Float16* arow = act + (size_t)(mt * 16 + mrow) * KFC + 8 * halfsel;
    const _Float16* wp   = wfc + (((size_t)nt * 1024) * 32 + lane) * 16;

    v8f c = {};
    #pragma unroll 8
    for (int kc = 0; kc < 1024; ++kc) {
        const _Float16* ap = arow + (size_t)kc * 32;
        v16h a;
        *(v8h*)&a       = *(const v8h*)(ap);
        *((v8h*)&a + 1) = *(const v8h*)(ap + 16);
        v16h bm = *(const v16h*)(wp + (size_t)kc * 512);
        c = __builtin_amdgcn_wmma_f32_16x16x32_f16(false, a, false, bm,
                                                   (short)0, c, false, false);
    }
#if defined(__HIP_DEVICE_COMPILE__)
    __builtin_amdgcn_s_wait_tensorcnt(0);
#endif
    const int n  = (nt << 4) + (lane & 15);
    const float bn = fcb[n];
    #pragma unroll
    for (int r = 0; r < 8; ++r) {
        const int row = (mt << 4) + r + (halfsel << 3);
        out[(size_t)row * NFC + n] = tanhf(c[r] + bn);
    }
}

// ---------------- host-side launch ----------------
extern "C" void kernel_launch(void* const* d_in, const int* in_sizes, int n_in,
                              void* d_out, int out_size, void* d_ws, size_t ws_size,
                              hipStream_t stream)
{
    // input order per setup_inputs() flattening:
    const float* x    = (const float*)d_in[0];
    const float* mean = (const float*)d_in[1];
    const float* eb0_w1 = (const float*)d_in[2];
    const float* eb0_b1 = (const float*)d_in[3];
    const float* eb0_w2 = (const float*)d_in[4];
    const float* eb0_b2 = (const float*)d_in[5];
    const float* eb0_ws = (const float*)d_in[6];
    const float* eb0_bs = (const float*)d_in[7];
    const float* eb1_w1 = (const float*)d_in[8];
    const float* eb1_b1 = (const float*)d_in[9];
    const float* eb1_w2 = (const float*)d_in[10];
    const float* eb1_b2 = (const float*)d_in[11];
    const float* eb2_w1 = (const float*)d_in[12];
    const float* eb2_b1 = (const float*)d_in[13];
    const float* eb2_w2 = (const float*)d_in[14];
    const float* eb2_b2 = (const float*)d_in[15];
    const float* eo_w   = (const float*)d_in[16];
    const float* eo_b   = (const float*)d_in[17];
    const float* di_w   = (const float*)d_in[18];
    const float* di_b   = (const float*)d_in[19];
    const float* db0_w1 = (const float*)d_in[20];
    const float* db0_b1 = (const float*)d_in[21];
    const float* db0_w2 = (const float*)d_in[22];
    const float* db0_b2 = (const float*)d_in[23];
    const float* db1_w1 = (const float*)d_in[24];
    const float* db1_b1 = (const float*)d_in[25];
    const float* db1_w2 = (const float*)d_in[26];
    const float* db1_b2 = (const float*)d_in[27];
    const float* db2_w1 = (const float*)d_in[28];
    const float* db2_b1 = (const float*)d_in[29];
    const float* db2_w2 = (const float*)d_in[30];
    const float* db2_b2 = (const float*)d_in[31];
    const float* fc_w   = (const float*)d_in[32];
    const float* fc_b   = (const float*)d_in[33];

    float* out = (float*)d_out;
    const size_t SHAT = (size_t)B_ * NFC;       // 393216 floats, then prob

    // workspace partition (bytes)
    const size_t ACT   = (size_t)B_ * L_ * CP * sizeof(_Float16);   // 50331648
    const size_t SLOTE = (size_t)KT_ * 2 * 4 * 32 * 16;             // 20480 f16 per conv slot
    char* ws = (char*)d_ws;
    _Float16* actA  = (_Float16*)(ws);
    _Float16* actB  = (_Float16*)(ws + ACT);
    _Float16* actC  = (_Float16*)(ws + 2 * ACT);
    _Float16* wconv = (_Float16*)(ws + 3 * ACT);
    _Float16* wfc   = (_Float16*)(ws + 3 * ACT + 15 * SLOTE * sizeof(_Float16));

    // ---- prep: pack weights ----
    auto packc = [&](int slot, const float* w, int cout, int cin, int kt) {
        int total = 4 * (kt * 2) * 32 * 16;
        pack_conv_w_kernel<<<(total + 255) / 256, 256, 0, stream>>>(
            w, wconv + (size_t)slot * SLOTE, cout, cin, kt);
    };
    packc( 0, eb0_w1, F_, 1,  KT_);
    packc( 1, eb0_w2, F_, F_, KT_);
    packc( 2, eb0_ws, F_, 1,  1);
    packc( 3, eb1_w1, F_, F_, KT_);
    packc( 4, eb1_w2, F_, F_, KT_);
    packc( 5, eb2_w1, F_, F_, KT_);
    packc( 6, eb2_w2, F_, F_, KT_);
    packc( 7, eo_w,   D_, F_, KT_);
    packc( 8, di_w,   F_, D_, KT_);
    packc( 9, db0_w1, F_, F_, KT_);
    packc(10, db0_w2, F_, F_, KT_);
    packc(11, db1_w1, F_, F_, KT_);
    packc(12, db1_w2, F_, F_, KT_);
    packc(13, db2_w1, F_, F_, KT_);
    packc(14, db2_w2, F_, F_, KT_);
    {
        int total = 32 * 1024 * 32 * 16;                    // 16777216
        pack_fc_w_kernel<<<total / 256, 256, 0, stream>>>(fc_w, wfc);
    }
    {
        int total = B_ * L_ * CP;
        pack_x_kernel<<<total / 256, 256, 0, stream>>>(x, actA);
    }

    // ---- network ----
    const int NT = B_ * (L_ / 16);                          // 24576 conv tiles
    auto w_of = [&](int slot) { return wconv + (size_t)slot * SLOTE; };

    // conv5: plain conv (bias), optional relu
    auto conv5 = [&](const _Float16* in, int slot, const float* bias,
                     _Float16* o, int relu, int cu) {
        if (relu)
            conv_wmma_kernel<KT_, false, true ><<<NT, 128, 0, stream>>>(in, w_of(slot), bias, nullptr, o, cu);
        else
            conv_wmma_kernel<KT_, false, false><<<NT, 128, 0, stream>>>(in, w_of(slot), bias, nullptr, o, cu);
    };
    // conv5s: conv + skip-add + relu
    auto conv5s = [&](const _Float16* in, int slot, const float* bias,
                      const _Float16* skip, _Float16* o, int cu) {
        conv_wmma_kernel<KT_, true, true><<<NT, 128, 0, stream>>>(in, w_of(slot), bias, skip, o, cu);
    };
    // conv1s: 1x1 conv + skip-add + relu
    auto conv1s = [&](const _Float16* in, int slot, const float* bias,
                      const _Float16* skip, _Float16* o, int cu) {
        conv_wmma_kernel<1, true, true><<<NT, 128, 0, stream>>>(in, w_of(slot), bias, skip, o, cu);
    };

    // enc_b0: t = relu(conv1(x)); h = conv2(t); out = relu(h + convS(x))
    conv5 (actA, 0, eb0_b1, actB, 1, F_);
    conv5 (actB, 1, eb0_b2, actC, 0, F_);
    conv1s(actA, 2, eb0_bs, actC, actB, F_);                // h in B
    // enc_b1
    conv5 (actB, 3, eb1_b1, actA, 1, F_);
    conv5s(actA, 4, eb1_b2, actB, actC, F_);                // h in C
    // enc_b2
    conv5 (actC, 5, eb2_b1, actA, 1, F_);
    conv5s(actA, 6, eb2_b2, actC, actB, F_);                // h in B
    // enc_out -> codes in A (D_ channels, rest zero)
    conv5 (actB, 7, eo_b, actA, 0, D_);

    // soft VQ: prob -> d_out[SHAT..], xq -> B
    code_assign_kernel<<<(B_ * L_) / 256, 256, 0, stream>>>(actA, mean, out + SHAT, actB);

    // dec_in (no relu)
    conv5 (actB, 8, di_b, actC, 0, F_);                     // s in C
    // dec_b0
    conv5 (actC,  9, db0_b1, actA, 1, F_);
    conv5s(actA, 10, db0_b2, actC, actB, F_);               // s in B
    // dec_b1
    conv5 (actB, 11, db1_b1, actA, 1, F_);
    conv5s(actA, 12, db1_b2, actB, actC, F_);               // s in C
    // dec_b2
    conv5 (actC, 13, db2_b1, actA, 1, F_);
    conv5s(actA, 14, db2_b2, actC, actB, F_);               // s in B

    // FC + tanh -> d_out[0..SHAT)
    fc_wmma_kernel<<<dim3(B_ / 16, NFC / 16), 32, 0, stream>>>(actB, wfc, fc_b, out);
}
